// VisionMamba_50105088475402
// MI455X (gfx1250) — matrix operand order repaired and checked by reading, not compiled, not run on hardware
//
#include <hip/hip_runtime.h>
#include <hip/hip_bf16.h>

// ---------------------------------------------------------------------------
// VisionMamba forward for gfx1250 (MI455X).
//  - big GEMMs: 8-wave workgroups, 64x64 tiles, double-buffered LDS staging
//    via GLOBAL_LOAD_ASYNC_TO_LDS_B128 (ASYNCcnt) + v_wmma_f32_16x16x32_bf16
//  - row-wise math (LN, conv1d, softplus, SSM scan, SiLU) stays f32 VALU
// ---------------------------------------------------------------------------

typedef __attribute__((ext_vector_type(16))) __bf16 v16bf;
typedef __attribute__((ext_vector_type(8)))  __bf16 v8bf;
typedef __attribute__((ext_vector_type(8)))  float  v8f;
typedef __attribute__((address_space(3)))    __bf16 lds_bf16;

#define BATCH   8
#define LTOK    197
#define DMODEL  768
#define DINNER  1536
#define DSTATE  16
#define DEPTH   12
#define MROWS   (BATCH * LTOK)      // 1576
#define MPAD    1600                // 25 * 64  (and 100 * 16)
#define PADROWS (MPAD - MROWS)      // 24

// ---------------------------------------------------------------------------
// LDS-tiled WMMA GEMM:  C[M x N] f32 = A[M x K] bf16 (row major)
//                                    x W[N x K] bf16 (row major, B = W^T)
// block = 256 threads (8 waves), tile = 64(M) x 64(N), K step 32.
// A/W tiles staged in LDS by per-lane async DMA (16B per lane), ping-pong
// buffered so the DMA of step k+1 overlaps the WMMAs of step k.
// grid = (M/64, N/64)
// ---------------------------------------------------------------------------
__global__ void __launch_bounds__(256)
wmma_gemm_lds(const __bf16* __restrict__ A,
              const __bf16* __restrict__ W,
              float* __restrict__ C,
              int K, int N)
{
    __shared__ alignas(32) __bf16 As[2][64 * 32];
    __shared__ alignas(32) __bf16 Ws[2][64 * 32];

    const int tid  = threadIdx.x;
    const int lane = tid & 31;
    const int wave = tid >> 5;          // 0..7
    const int wm   = wave & 3;          // M subtile (16 rows) within 64
    const int wn   = wave >> 2;         // N half (32 cols) within 64
    const int hl   = lane >> 4;         // 0 = lanes 0-15, 1 = lanes 16-31
    const int l16  = lane & 15;

    const int tm = blockIdx.x;
    const int tn = blockIdx.y;

    // staging assignment: 256 lanes = 64 rows x 4 chunks of 8 bf16 (16B)
    const int srow = tid >> 2;          // 0..63
    const int schk = tid & 3;           // 0..3
    const __bf16* agp = A + (size_t)(tm * 64 + srow) * (size_t)K + schk * 8;
    const __bf16* wgp = W + (size_t)(tn * 64 + srow) * (size_t)K + schk * 8;

    lds_bf16* asl[2] = { (lds_bf16*)&As[0][srow * 32 + schk * 8],
                         (lds_bf16*)&As[1][srow * 32 + schk * 8] };
    lds_bf16* wsl[2] = { (lds_bf16*)&Ws[0][srow * 32 + schk * 8],
                         (lds_bf16*)&Ws[1][srow * 32 + schk * 8] };

    // prologue: stage k0 = 0 into buffer 0
    {
        unsigned long long ga = (unsigned long long)(uintptr_t)agp;
        unsigned long long gw = (unsigned long long)(uintptr_t)wgp;
        asm volatile("global_load_async_to_lds_b128 %0, %1, off"
                     :: "v"(asl[0]), "v"(ga) : "memory");
        asm volatile("global_load_async_to_lds_b128 %0, %1, off"
                     :: "v"(wsl[0]), "v"(gw) : "memory");
    }

    v8f acc0 = {}, acc1 = {};
    int cur = 0;
    for (int k0 = 0; k0 < K; k0 += 32) {
        // drain my DMA lanes, then make every wave's staging visible
        asm volatile("s_wait_asynccnt 0" ::: "memory");
        __syncthreads();

        // overlap: kick off next K-step's DMA into the other buffer
        if (k0 + 32 < K) {
            unsigned long long ga = (unsigned long long)(uintptr_t)(agp + k0 + 32);
            unsigned long long gw = (unsigned long long)(uintptr_t)(wgp + k0 + 32);
            asm volatile("global_load_async_to_lds_b128 %0, %1, off"
                         :: "v"(asl[cur ^ 1]), "v"(ga) : "memory");
            asm volatile("global_load_async_to_lds_b128 %0, %1, off"
                         :: "v"(wsl[cur ^ 1]), "v"(gw) : "memory");
        }

        // A fragment (16x32): lane<16 -> K {0..15}; lane>=16 -> {16..31}
        const v8bf alo = *(const v8bf*)&As[cur][(wm * 16 + l16) * 32 + hl * 16];
        const v8bf ahi = *(const v8bf*)&As[cur][(wm * 16 + l16) * 32 + hl * 16 + 8];
        v16bf a;
#pragma unroll
        for (int i = 0; i < 8; ++i) { a[i] = alo[i]; a[i + 8] = ahi[i]; }

        // two B fragments (32x16 each): column = lane&15, K half by lane group
        const v16bf b0 = *(const v16bf*)&Ws[cur][(wn * 32 + l16) * 32 + hl * 16];
        const v16bf b1 = *(const v16bf*)&Ws[cur][(wn * 32 + 16 + l16) * 32 + hl * 16];

        acc0 = __builtin_amdgcn_wmma_f32_16x16x32_bf16(false, a, false, b0,
                                                       (short)0, acc0, false, false);
        acc1 = __builtin_amdgcn_wmma_f32_16x16x32_bf16(false, a, false, b1,
                                                       (short)0, acc1, false, false);
        cur ^= 1;
    }

    // D layout: row = base + hl*8 + v ; col = base + (lane&15)
    const int crow = tm * 64 + wm * 16 + hl * 8;
    const int ccol = tn * 64 + wn * 32 + l16;
    float* c0 = C + (size_t)crow * (size_t)N + ccol;
#pragma unroll
    for (int v = 0; v < 8; ++v) {
        c0[(size_t)v * N]      = acc0[v];
        c0[(size_t)v * N + 16] = acc1[v];
    }
}

// ---------------------------------------------------------------------------
// Single-wave WMMA GEMM (for the tiny N=32 x_proj GEMM). grid = (M/16, N/16)
// ---------------------------------------------------------------------------
__global__ void wmma_gemm_small(const __bf16* __restrict__ A,
                                const __bf16* __restrict__ W,
                                float* __restrict__ C,
                                int K, int N)
{
    const int tm   = blockIdx.x;
    const int tn   = blockIdx.y;
    const int lane = threadIdx.x & 31;
    const int hl   = lane >> 4;
    const int l16  = lane & 15;

    const __bf16* arow = A + (size_t)(tm * 16 + l16) * (size_t)K;
    const __bf16* wrow = W + (size_t)(tn * 16 + l16) * (size_t)K;

    v8f acc = {};
    for (int k0 = 0; k0 < K; k0 += 32) {
        const v8bf alo = *reinterpret_cast<const v8bf*>(arow + k0 + hl * 16);
        const v8bf ahi = *reinterpret_cast<const v8bf*>(arow + k0 + hl * 16 + 8);
        v16bf a;
#pragma unroll
        for (int i = 0; i < 8; ++i) { a[i] = alo[i]; a[i + 8] = ahi[i]; }
        const v16bf b = *reinterpret_cast<const v16bf*>(wrow + k0 + hl * 16);
        acc = __builtin_amdgcn_wmma_f32_16x16x32_bf16(false, a, false, b,
                                                      (short)0, acc, false, false);
    }
    float* crow = C + (size_t)(tm * 16 + hl * 8) * (size_t)N + tn * 16 + l16;
#pragma unroll
    for (int v = 0; v < 8; ++v) crow[(size_t)v * N] = acc[v];
}

// ---------------------------------------------------------------------------
// Patch embed + cls token + pos embed  ->  t[MROWS x DMODEL] (f32)
// ---------------------------------------------------------------------------
__global__ void patch_embed_kernel(const float* __restrict__ x,
                                   const float* __restrict__ cw,
                                   const float* __restrict__ cb,
                                   const float* __restrict__ cls,
                                   const float* __restrict__ pos,
                                   float* __restrict__ t)
{
    int idx = blockIdx.x * blockDim.x + threadIdx.x;
    if (idx >= MROWS * DMODEL) return;
    int d   = idx % DMODEL;
    int row = idx / DMODEL;
    int b   = row / LTOK;
    int l   = row % LTOK;

    float v;
    if (l == 0) {
        v = cls[d];
    } else {
        int p  = l - 1;
        int py = p / 14, px = p % 14;
        const float* xb = x + (size_t)b * 3 * 224 * 224;
        const float* wd = cw + (size_t)d * (3 * 16 * 16);
        float acc = cb[d];
        for (int c = 0; c < 3; ++c)
            for (int ky = 0; ky < 16; ++ky) {
                const float* xr = xb + ((size_t)c * 224 + (py * 16 + ky)) * 224 + px * 16;
                const float* wr = wd + (c * 16 + ky) * 16;
#pragma unroll
                for (int kx = 0; kx < 16; ++kx) acc += xr[kx] * wr[kx];
            }
        v = acc;
    }
    t[idx] = v + pos[l * DMODEL + d];
}

// ---------------------------------------------------------------------------
// LayerNorm (f32 in) -> bf16 out, pad rows -> 0.  One block (256) per row.
// ---------------------------------------------------------------------------
__global__ void ln_to_bf16_kernel(const float* __restrict__ t,
                                  const float* __restrict__ w,
                                  const float* __restrict__ b,
                                  __bf16* __restrict__ o)
{
    const int row = blockIdx.x;
    __bf16* orow = o + (size_t)row * DMODEL;
    if (row >= MROWS) {
        for (int i = threadIdx.x; i < DMODEL; i += blockDim.x) orow[i] = (__bf16)0.0f;
        return;
    }
    const float* tr = t + (size_t)row * DMODEL;
    float s = 0.0f, s2 = 0.0f;
    for (int i = threadIdx.x; i < DMODEL; i += 256) {
        float v = tr[i]; s += v; s2 += v * v;
    }
    __shared__ float r1[256], r2[256];
    r1[threadIdx.x] = s; r2[threadIdx.x] = s2;
    __syncthreads();
    for (int st = 128; st > 0; st >>= 1) {
        if (threadIdx.x < st) {
            r1[threadIdx.x] += r1[threadIdx.x + st];
            r2[threadIdx.x] += r2[threadIdx.x + st];
        }
        __syncthreads();
    }
    const float m   = r1[0] * (1.0f / DMODEL);
    const float var = r2[0] * (1.0f / DMODEL) - m * m;
    const float rs  = rsqrtf(var + 1e-5f);
    for (int i = threadIdx.x; i < DMODEL; i += 256)
        orow[i] = (__bf16)((tr[i] - m) * rs * w[i] + b[i]);
}

// Final LayerNorm -> f32 output
__global__ void ln_out_kernel(const float* __restrict__ t,
                              const float* __restrict__ w,
                              const float* __restrict__ b,
                              float* __restrict__ o)
{
    const int row = blockIdx.x;
    const float* tr = t + (size_t)row * DMODEL;
    float s = 0.0f, s2 = 0.0f;
    for (int i = threadIdx.x; i < DMODEL; i += 256) {
        float v = tr[i]; s += v; s2 += v * v;
    }
    __shared__ float r1[256], r2[256];
    r1[threadIdx.x] = s; r2[threadIdx.x] = s2;
    __syncthreads();
    for (int st = 128; st > 0; st >>= 1) {
        if (threadIdx.x < st) {
            r1[threadIdx.x] += r1[threadIdx.x + st];
            r2[threadIdx.x] += r2[threadIdx.x + st];
        }
        __syncthreads();
    }
    const float m   = r1[0] * (1.0f / DMODEL);
    const float var = r2[0] * (1.0f / DMODEL) - m * m;
    const float rs  = rsqrtf(var + 1e-5f);
    for (int i = threadIdx.x; i < DMODEL; i += 256)
        o[(size_t)row * DMODEL + i] = (tr[i] - m) * rs * w[i] + b[i];
}

// ---------------------------------------------------------------------------
// Depthwise causal conv1d (k=4) + SiLU on xi = xz[:, :DINNER]
// ---------------------------------------------------------------------------
__global__ void conv1d_silu_kernel(const float* __restrict__ xz,
                                   const float* __restrict__ cw,
                                   const float* __restrict__ cb,
                                   float* __restrict__ xc,
                                   __bf16* __restrict__ xcbf)
{
    int idx = blockIdx.x * blockDim.x + threadIdx.x;
    if (idx >= MPAD * DINNER) return;
    int e   = idx % DINNER;
    int row = idx / DINNER;
    if (row >= MROWS) { xc[idx] = 0.0f; xcbf[idx] = (__bf16)0.0f; return; }
    int b = row / LTOK, l = row % LTOK;

    float acc = cb[e];
#pragma unroll
    for (int j = 0; j < 4; ++j) {
        int ls = l - 3 + j;
        if (ls >= 0)
            acc += cw[e * 4 + j] * xz[(size_t)(b * LTOK + ls) * (2 * DINNER) + e];
    }
    float s = acc / (1.0f + expf(-acc));   // SiLU
    xc[idx]   = s;
    xcbf[idx] = (__bf16)s;
}

// delta = softplus(raw + dt_b), in place
__global__ void softplus_bias_kernel(float* __restrict__ dlt,
                                     const float* __restrict__ dtb)
{
    int idx = blockIdx.x * blockDim.x + threadIdx.x;
    if (idx >= MPAD * DINNER) return;
    int e = idx % DINNER;
    float v = dlt[idx] + dtb[e];
    dlt[idx] = (v > 20.0f) ? v : log1pf(expf(v));
}

// ---------------------------------------------------------------------------
// Sequential SSM scan: one thread per (batch, channel), 16 f32 states in
// registers, 197 steps. Fuses + D*xc and * silu(z); writes bf16 y.
// ---------------------------------------------------------------------------
__global__ void ssm_scan_kernel(const float* __restrict__ dlt,
                                const float* __restrict__ xc,
                                const float* __restrict__ xdbl,
                                const float* __restrict__ xz,
                                const float* __restrict__ A_log,
                                const float* __restrict__ Dp,
                                __bf16* __restrict__ ybf)
{
    int tid = blockIdx.x * blockDim.x + threadIdx.x;
    if (tid >= BATCH * DINNER) return;
    int b = tid / DINNER;
    int e = tid % DINNER;

    float As[DSTATE];
#pragma unroll
    for (int s = 0; s < DSTATE; ++s) As[s] = -expf(A_log[e * DSTATE + s]);
    const float dpe = Dp[e];

    float st[DSTATE];
#pragma unroll
    for (int s = 0; s < DSTATE; ++s) st[s] = 0.0f;

    for (int l = 0; l < LTOK; ++l) {
        const size_t row = (size_t)b * LTOK + l;
        const float d  = dlt[row * DINNER + e];
        const float xv = xc[row * DINNER + e];
        const float dx = d * xv;
        const float* Bp = xdbl + row * (2 * DSTATE) + DSTATE; // B_ssm = x_dbl[..,16:32]
        float acc = 0.0f;
#pragma unroll
        for (int s = 0; s < DSTATE; ++s) {
            const float Bv = Bp[s];
            st[s] = expf(d * As[s]) * st[s] + dx * Bv;
            acc  += st[s] * Bv;
        }
        float y  = acc + dpe * xv;
        float zv = xz[row * (2 * DINNER) + DINNER + e];
        y *= zv / (1.0f + expf(-zv));                 // * silu(z)
        ybf[row * DINNER + e] = (__bf16)y;
    }
}

// ---------------------------------------------------------------------------
// Small helpers
// ---------------------------------------------------------------------------
__global__ void f32_to_bf16_kernel(const float* __restrict__ a,
                                   __bf16* __restrict__ o, int n)
{
    int i = blockIdx.x * blockDim.x + threadIdx.x;
    if (i < n) o[i] = (__bf16)a[i];
}

__global__ void zero_bf16_kernel(__bf16* __restrict__ o, int n)
{
    int i = blockIdx.x * blockDim.x + threadIdx.x;
    if (i < n) o[i] = (__bf16)0.0f;
}

__global__ void residual_add_kernel(float* __restrict__ t,
                                    const float* __restrict__ o, int n)
{
    int i = blockIdx.x * blockDim.x + threadIdx.x;
    if (i < n) t[i] += o[i];
}

// ---------------------------------------------------------------------------
// Orchestration
// ---------------------------------------------------------------------------
extern "C" void kernel_launch(void* const* d_in, const int* in_sizes, int n_in,
                              void* d_out, int out_size, void* d_ws, size_t ws_size,
                              hipStream_t stream)
{
    (void)in_sizes; (void)n_in; (void)out_size; (void)ws_size;

    const float* x        = (const float*)d_in[0];
    const float* conv_w   = (const float*)d_in[1];
    const float* conv_b   = (const float*)d_in[2];
    const float* cls_tok  = (const float*)d_in[3];
    const float* pos_emb  = (const float*)d_in[4];
    const float* ln_w     = (const float*)d_in[5];
    const float* ln_b     = (const float*)d_in[6];
    const float* in_proj  = (const float*)d_in[7];
    const float* c1w      = (const float*)d_in[8];
    const float* c1b      = (const float*)d_in[9];
    const float* xp_w     = (const float*)d_in[10];
    const float* dt_w     = (const float*)d_in[11];
    const float* dt_b     = (const float*)d_in[12];
    const float* A_log    = (const float*)d_in[13];
    const float* D_param  = (const float*)d_in[14];
    const float* out_w    = (const float*)d_in[15];
    const float* norm_w   = (const float*)d_in[16];
    const float* norm_b   = (const float*)d_in[17];

    char*  ws  = (char*)d_ws;
    size_t off = 0;
    auto take = [&](size_t bytes) -> char* {
        char* p = ws + off;
        off += (bytes + 255) & ~(size_t)255;
        return p;
    };

    float*  t     = (float*) take((size_t)MROWS * DMODEL * 4);
    __bf16* xln   = (__bf16*)take((size_t)MPAD * DMODEL * 2);
    float*  xz    = (float*) take((size_t)MPAD * 2 * DINNER * 4);
    float*  xc    = (float*) take((size_t)MPAD * DINNER * 4);
    __bf16* xcbf  = (__bf16*)take((size_t)MPAD * DINNER * 2);
    float*  dlt   = (float*) take((size_t)MPAD * DINNER * 4);
    float*  xdbl  = (float*) take((size_t)MPAD * 2 * DSTATE * 4);
    __bf16* ybf   = (__bf16*)take((size_t)MPAD * DINNER * 2);
    float*  obuf  = (float*) take((size_t)MPAD * DMODEL * 4);
    __bf16* w_in  = (__bf16*)take((size_t)2 * DINNER * DMODEL * 2);
    __bf16* w_dt  = (__bf16*)take((size_t)DINNER * DINNER * 2);
    __bf16* w_xp  = (__bf16*)take((size_t)2 * DSTATE * DINNER * 2);
    __bf16* w_out = (__bf16*)take((size_t)DMODEL * DINNER * 2);

    const int TB = 256;

    // pad rows of the scan output are never written by the scan -> zero them
    {
        int n = PADROWS * DINNER;
        zero_bf16_kernel<<<(n + TB - 1) / TB, TB, 0, stream>>>(ybf + (size_t)MROWS * DINNER, n);
    }

    // patch embedding + cls + pos
    {
        int n = MROWS * DMODEL;
        patch_embed_kernel<<<(n + TB - 1) / TB, TB, 0, stream>>>(x, conv_w, conv_b, cls_tok, pos_emb, t);
    }

    for (int i = 0; i < DEPTH; ++i) {
        // ---- per-layer weight -> bf16 ----
        {
            int n = 2 * DINNER * DMODEL;
            f32_to_bf16_kernel<<<(n + TB - 1) / TB, TB, 0, stream>>>(in_proj + (size_t)i * n, w_in, n);
        }
        {
            int n = DINNER * DINNER;
            f32_to_bf16_kernel<<<(n + TB - 1) / TB, TB, 0, stream>>>(dt_w + (size_t)i * n, w_dt, n);
        }
        {
            int n = 2 * DSTATE * DINNER;
            f32_to_bf16_kernel<<<(n + TB - 1) / TB, TB, 0, stream>>>(xp_w + (size_t)i * n, w_xp, n);
        }
        {
            int n = DMODEL * DINNER;
            f32_to_bf16_kernel<<<(n + TB - 1) / TB, TB, 0, stream>>>(out_w + (size_t)i * n, w_out, n);
        }

        // ---- layernorm -> bf16 activations ----
        ln_to_bf16_kernel<<<MPAD, TB, 0, stream>>>(t, ln_w + (size_t)i * DMODEL,
                                                   ln_b + (size_t)i * DMODEL, xln);

        // ---- xz = xln @ in_proj^T  (M x 3072, K=768) ----
        wmma_gemm_lds<<<dim3(MPAD / 64, (2 * DINNER) / 64), 256, 0, stream>>>(
            xln, w_in, xz, DMODEL, 2 * DINNER);

        // ---- depthwise causal conv1d + SiLU ----
        {
            int n = MPAD * DINNER;
            conv1d_silu_kernel<<<(n + TB - 1) / TB, TB, 0, stream>>>(
                xz, c1w + (size_t)i * DINNER * 4, c1b + (size_t)i * DINNER, xc, xcbf);
        }

        // ---- delta_raw = xc @ dt_proj^T  (M x 1536, K=1536) ----
        wmma_gemm_lds<<<dim3(MPAD / 64, DINNER / 64), 256, 0, stream>>>(
            xcbf, w_dt, dlt, DINNER, DINNER);

        // ---- delta = softplus(delta_raw + dt_b) ----
        {
            int n = MPAD * DINNER;
            softplus_bias_kernel<<<(n + TB - 1) / TB, TB, 0, stream>>>(
                dlt, dt_b + (size_t)i * DINNER);
        }

        // ---- x_dbl = xc @ x_proj^T  (M x 32, K=1536), tiny N: single-wave ----
        wmma_gemm_small<<<dim3(MPAD / 16, (2 * DSTATE) / 16), 32, 0, stream>>>(
            xcbf, w_xp, xdbl, DINNER, 2 * DSTATE);

        // ---- sequential selective scan (fused D*xc and silu(z) gate) ----
        {
            int n = BATCH * DINNER;
            ssm_scan_kernel<<<(n + TB - 1) / TB, TB, 0, stream>>>(
                dlt, xc, xdbl, xz, A_log + (size_t)i * DINNER * DSTATE,
                D_param + (size_t)i * DINNER, ybf);
        }

        // ---- out = y @ out_proj^T  (M x 768, K=1536) ----
        wmma_gemm_lds<<<dim3(MPAD / 64, DMODEL / 64), 256, 0, stream>>>(
            ybf, w_out, obuf, DINNER, DMODEL);

        // ---- residual ----
        {
            int n = MROWS * DMODEL;
            residual_add_kernel<<<(n + TB - 1) / TB, TB, 0, stream>>>(t, obuf, n);
        }
    }

    // final layernorm -> f32 output
    ln_out_kernel<<<MROWS, TB, 0, stream>>>(t, norm_w, norm_b, (float*)d_out);
}
